// GroupMessagePassing_70342974374333
// MI455X (gfx1250) — matrix operand order repaired and checked by reading, not compiled, not run on hardware
//
#include <hip/hip_runtime.h>

#define NN 1024
#define DD 128
#define HH 256
#define TI 8

typedef __attribute__((ext_vector_type(16))) __bf16 v16bf;
typedef __attribute__((ext_vector_type(8)))  float  v8f;
typedef __attribute__((ext_vector_type(4)))  unsigned int v4u;

__device__ __forceinline__ unsigned short f32_to_bf16_rne(float f) {
  unsigned int u = __float_as_uint(f);
  u += 0x7fffu + ((u >> 16) & 1u);   // round-to-nearest-even
  return (unsigned short)(u >> 16);
}

__global__ void cvt_bf16_kernel(const float* __restrict__ src,
                                unsigned short* __restrict__ dst, int n) {
  int i = blockIdx.x * blockDim.x + threadIdx.x;
  if (i < n) dst[i] = f32_to_bf16_rne(src[i]);
}

// dst[r*cols+c] = bf16(src[r*src_stride + src_off + c])   (splits W1 into W1a/W1b)
__global__ void cvt_bf16_strided_kernel(const float* __restrict__ src,
                                        unsigned short* __restrict__ dst,
                                        int rows, int cols, int src_stride, int src_off) {
  int i = blockIdx.x * blockDim.x + threadIdx.x;
  if (i >= rows * cols) return;
  int r = i / cols, c = i - r * cols;
  dst[i] = f32_to_bf16_rne(src[(size_t)r * src_stride + src_off + c]);
}

// C[M x Ncols] = A[M x K](bf16) @ W[Ncols x K](bf16)^T + bias[col]*rowscale[row]
// One wave computes one 16x16 tile with v_wmma_f32_16x16x32_bf16, K-loop step 32.
__global__ void wmma_gemm_kernel(const unsigned short* __restrict__ A,
                                 const unsigned short* __restrict__ W,
                                 const float* __restrict__ bias,
                                 const float* __restrict__ rowscale,
                                 float* __restrict__ C,
                                 int M, int Ncols, int K) {
  const int lane  = threadIdx.x & 31;
  const int wave  = threadIdx.x >> 5;
  const int tile  = blockIdx.x * (blockDim.x >> 5) + wave;
  const int tilesN = Ncols >> 4;
  if (tile >= (M >> 4) * tilesN) return;          // uniform per wave: EXEC stays all-1s
  const int tm   = (tile / tilesN) << 4;
  const int tn   = (tile - (tile / tilesN) * tilesN) << 4;
  const int half = lane >> 4;
  const int l15  = lane & 15;

  v8f acc;
  {
    float b = bias ? bias[tn + l15] : 0.0f;
    #pragma unroll
    for (int r = 0; r < 8; ++r) {
      float rs = rowscale ? rowscale[tm + r + 8 * half] : 1.0f;
      acc[r] = b * rs;
    }
  }

  const unsigned short* Arow = A + (size_t)(tm + l15) * K;
  const unsigned short* Wrow = W + (size_t)(tn + l15) * K;

  for (int k0 = 0; k0 < K; k0 += 32) {
    union { v16bf v; v4u q[2]; } ua, ub;
    const v4u* pa = (const v4u*)(Arow + k0);
    const v4u* pb = (const v4u*)(Wrow + k0);
    // A 16x32 bf16 (ISA 7.12.2): lane-half h holds K {8h..8h+7} U {16+8h..16+8h+7}
    ua.q[0] = pa[half];
    ua.q[1] = pa[2 + half];
    // B 32x16 bf16: column N=l15 per lane; lane-half h holds K {8h..8h+7} U {16+8h..16+8h+7}
    ub.q[0] = pb[half];
    ub.q[1] = pb[2 + half];
    acc = __builtin_amdgcn_wmma_f32_16x16x32_bf16(
        /*neg_a=*/false, ua.v, /*neg_b=*/false, ub.v,
        /*c_mod=*/(short)0, acc, /*reuse_a=*/false, /*reuse_b=*/false);
  }

  #pragma unroll
  for (int r = 0; r < 8; ++r)
    C[(size_t)(tm + r + 8 * half) * Ncols + (tn + l15)] = acc[r];
}

// deg[i] = #neighbors of row i
__global__ void deg_kernel(const int* __restrict__ adj, float* __restrict__ deg) {
  __shared__ float red[256];
  const int i = blockIdx.x;
  float s = 0.0f;
  for (int j = threadIdx.x; j < NN; j += 256)
    s += (adj[(size_t)i * NN + j] > 0) ? 1.0f : 0.0f;
  red[threadIdx.x] = s;
  __syncthreads();
  for (int w = 128; w > 0; w >>= 1) {
    if (threadIdx.x < w) red[threadIdx.x] += red[threadIdx.x + w];
    __syncthreads();
  }
  if (threadIdx.x == 0) deg[i] = red[0];
}

// agg[i,h] = sum_j mask[i,j] * relu(src[i,h] + nbr[j,h])  — fused, never materializes [N,N,H]
// Block: 8 i-rows x 256 h-lanes; mask row tile staged in LDS (broadcast reads).
__global__ void agg_kernel(const float* __restrict__ src, const float* __restrict__ nbr,
                           const int* __restrict__ adj, float* __restrict__ agg) {
  __shared__ float smask[TI * NN];   // 32 KB
  const int i0 = blockIdx.x * TI;
  const int h  = threadIdx.x;        // blockDim.x == HH == 256
  for (int idx = threadIdx.x; idx < TI * NN; idx += HH) {
    int t = idx >> 10;               // / NN
    int j = idx & (NN - 1);
    smask[idx] = (adj[(size_t)(i0 + t) * NN + j] > 0) ? 1.0f : 0.0f;
  }
  __syncthreads();
  float s[TI], acc[TI];
  #pragma unroll
  for (int t = 0; t < TI; ++t) {
    s[t] = src[(size_t)(i0 + t) * HH + h];   // b1 already folded in
    acc[t] = 0.0f;
  }
  #pragma unroll 4
  for (int j = 0; j < NN; ++j) {
    float nv = nbr[(size_t)j * HH + h];
    #pragma unroll
    for (int t = 0; t < TI; ++t)
      acc[t] = fmaf(smask[t * NN + j], fmaxf(s[t] + nv, 0.0f), acc[t]);
  }
  #pragma unroll
  for (int t = 0; t < TI; ++t)
    agg[(size_t)(i0 + t) * HH + h] = acc[t];
}

// GRU cell pointwise: out = (1-z)*n + z*x
__global__ void gru_kernel(const float* __restrict__ gi, const float* __restrict__ gh,
                           const float* __restrict__ x, float* __restrict__ out) {
  int idx = blockIdx.x * blockDim.x + threadIdx.x;
  if (idx >= NN * DD) return;
  int row = idx >> 7;                // / DD
  int col = idx & (DD - 1);
  const float* gir = gi + (size_t)row * 3 * DD;
  const float* ghr = gh + (size_t)row * 3 * DD;
  float r = 1.0f / (1.0f + __expf(-(gir[col] + ghr[col])));
  float z = 1.0f / (1.0f + __expf(-(gir[DD + col] + ghr[DD + col])));
  float n = tanhf(gir[2 * DD + col] + r * ghr[2 * DD + col]);
  out[idx] = (1.0f - z) * n + z * x[idx];
}

extern "C" void kernel_launch(void* const* d_in, const int* in_sizes, int n_in,
                              void* d_out, int out_size, void* d_ws, size_t ws_size,
                              hipStream_t stream) {
  (void)in_sizes; (void)n_in; (void)out_size; (void)ws_size;
  const float* X   = (const float*)d_in[0];
  const int*   adj = (const int*)  d_in[1];
  const float* W1  = (const float*)d_in[2];
  const float* b1  = (const float*)d_in[3];
  const float* W2  = (const float*)d_in[4];
  const float* b2  = (const float*)d_in[5];
  const float* Wih = (const float*)d_in[6];
  const float* Whh = (const float*)d_in[7];
  const float* bih = (const float*)d_in[8];
  const float* bhh = (const float*)d_in[9];
  float* out = (float*)d_out;

  char* ws = (char*)d_ws;
  size_t off = 0;
  auto alloc = [&](size_t bytes) -> void* {
    off = (off + 255) & ~(size_t)255;
    void* p = ws + off;
    off += bytes;
    return p;
  };

  unsigned short* Xb   = (unsigned short*)alloc((size_t)NN * DD * 2);
  unsigned short* W1a  = (unsigned short*)alloc((size_t)HH * DD * 2);
  unsigned short* W1b  = (unsigned short*)alloc((size_t)HH * DD * 2);
  unsigned short* W2b  = (unsigned short*)alloc((size_t)DD * HH * 2);
  unsigned short* Wihb = (unsigned short*)alloc((size_t)3 * DD * DD * 2);
  unsigned short* Whhb = (unsigned short*)alloc((size_t)3 * DD * DD * 2);
  float* srcb = (float*)alloc((size_t)NN * HH * 4);
  float* nbrb = (float*)alloc((size_t)NN * HH * 4);
  float* deg  = (float*)alloc((size_t)NN * 4);
  float* agg  = (float*)alloc((size_t)NN * HH * 4);
  unsigned short* aggb = (unsigned short*)alloc((size_t)NN * HH * 2);
  float* msg  = (float*)alloc((size_t)NN * DD * 4);
  unsigned short* msgb = (unsigned short*)alloc((size_t)NN * DD * 2);
  float* gi   = (float*)alloc((size_t)NN * 3 * DD * 4);
  float* gh   = (float*)alloc((size_t)NN * 3 * DD * 4);

  // fp32 -> bf16 conversions
  cvt_bf16_kernel<<<(NN * DD + 255) / 256, 256, 0, stream>>>(X, Xb, NN * DD);
  cvt_bf16_strided_kernel<<<(HH * DD + 255) / 256, 256, 0, stream>>>(W1, W1a, HH, DD, 2 * DD, 0);
  cvt_bf16_strided_kernel<<<(HH * DD + 255) / 256, 256, 0, stream>>>(W1, W1b, HH, DD, 2 * DD, DD);
  cvt_bf16_kernel<<<(DD * HH + 255) / 256, 256, 0, stream>>>(W2, W2b, DD * HH);
  cvt_bf16_kernel<<<(3 * DD * DD + 255) / 256, 256, 0, stream>>>(Wih, Wihb, 3 * DD * DD);
  cvt_bf16_kernel<<<(3 * DD * DD + 255) / 256, 256, 0, stream>>>(Whh, Whhb, 3 * DD * DD);

  // src = X@W1a^T + b1 ; nbr = X@W1b^T     (1024x256, K=128) -> 1024 tiles / 8 waves
  wmma_gemm_kernel<<<(NN / 16) * (HH / 16) / 8, 256, 0, stream>>>(Xb, W1a, b1, nullptr, srcb, NN, HH, DD);
  wmma_gemm_kernel<<<(NN / 16) * (HH / 16) / 8, 256, 0, stream>>>(Xb, W1b, nullptr, nullptr, nbrb, NN, HH, DD);

  deg_kernel<<<NN, 256, 0, stream>>>(adj, deg);
  agg_kernel<<<NN / TI, HH, 0, stream>>>(srcb, nbrb, adj, agg);

  // messages = agg@W2^T + deg*b2           (1024x128, K=256)
  cvt_bf16_kernel<<<(NN * HH + 255) / 256, 256, 0, stream>>>(agg, aggb, NN * HH);
  wmma_gemm_kernel<<<(NN / 16) * (DD / 16) / 8, 256, 0, stream>>>(aggb, W2b, b2, deg, msg, NN, DD, HH);

  // gi = messages@Wih^T + b_ih ; gh = X@Whh^T + b_hh   (1024x384, K=128)
  cvt_bf16_kernel<<<(NN * DD + 255) / 256, 256, 0, stream>>>(msg, msgb, NN * DD);
  wmma_gemm_kernel<<<(NN / 16) * (3 * DD / 16) / 8, 256, 0, stream>>>(msgb, Wihb, bih, nullptr, gi, NN, 3 * DD, DD);
  wmma_gemm_kernel<<<(NN / 16) * (3 * DD / 16) / 8, 256, 0, stream>>>(Xb, Whhb, bhh, nullptr, gh, NN, 3 * DD, DD);

  gru_kernel<<<(NN * DD + 255) / 256, 256, 0, stream>>>(gi, gh, X, out);
}
